// GAT2_63239098466923
// MI455X (gfx1250) — compile-verified, matrix operand order checked by reading, not run on hardware
//
#include <hip/hip_runtime.h>
#include <hip/hip_bf16.h>

typedef __attribute__((ext_vector_type(16))) __bf16 v16bf;
typedef __attribute__((ext_vector_type(8)))  float  v8f;

#define NEG_SLOPE 0.2f

static __device__ __forceinline__ unsigned fenc(float f) {
    unsigned u = __float_as_uint(f);
    return (u & 0x80000000u) ? ~u : (u | 0x80000000u);
}
static __device__ __forceinline__ float fdec(unsigned u) {
    unsigned v = (u & 0x80000000u) ? (u & 0x7fffffffu) : ~u;
    return __uint_as_float(v);
}
static __device__ __forceinline__ float lrelu(float x) { return x > 0.f ? x : NEG_SLOPE * x; }

// ---------------------------------------------------------------------------
// Zero a u32 range
__global__ void k_zero(unsigned* p, int n) {
    int i = blockIdx.x * blockDim.x + threadIdx.x;
    if (i < n) p[i] = 0u;
}

// ---------------------------------------------------------------------------
// Pack layer-1 weights [W_l1 | W_r1] (256x64 each) into WMMA bf16 B-layout.
// Bp[((kk*8+ct)*32 + lane)*16 + j],  K = kk*32 + (j<8 ? 8*kg+j : 16+8*kg+(j-8))
__global__ void k_pack1(const float* __restrict__ Wl, const float* __restrict__ Wr,
                        __bf16* __restrict__ Bp) {
    int id = blockIdx.x * blockDim.x + threadIdx.x;
    if (id >= 8 * 8 * 32 * 16) return;
    int j = id & 15, lane = (id >> 4) & 31, ct = (id >> 9) & 7, kk = (id >> 12) & 7;
    int kg = lane >> 4, n = lane & 15;
    int kloc = (j < 8) ? (kg * 8 + j) : (16 + kg * 8 + (j - 8));
    int K = kk * 32 + kloc;
    float v = (ct < 4) ? Wl[K * 64 + ct * 16 + n] : Wr[K * 64 + (ct - 4) * 16 + n];
    Bp[id] = (__bf16)v;
}

// Pack layer-2 weights [W_l2 | W_r2] (64x40 each, zero-padded to 48 cols -> 6 tiles)
__global__ void k_pack2(const float* __restrict__ Wl, const float* __restrict__ Wr,
                        __bf16* __restrict__ Bp) {
    int id = blockIdx.x * blockDim.x + threadIdx.x;
    if (id >= 2 * 6 * 32 * 16) return;
    int j = id & 15;
    int rest = id >> 4;
    int lane = rest & 31;
    int rest2 = rest >> 5;
    int ct = rest2 % 6, kk = rest2 / 6;
    int kg = lane >> 4, n = lane & 15;
    int kloc = (j < 8) ? (kg * 8 + j) : (16 + kg * 8 + (j - 8));
    int K = kk * 32 + kloc;
    int col = (ct % 3) * 16 + n;
    float v = 0.f;
    if (col < 40) v = (ct < 3) ? Wl[K * 40 + col] : Wr[K * 40 + col];
    Bp[id] = (__bf16)v;
}

// ---------------------------------------------------------------------------
// GEMM1: x(N x 256) @ packed(256 x 128) -> xl1(N x 64), xr1(N x 64)  (+bias)
// One wave per 16-row tile; 8 column tiles; K = 256 in 8 WMMA k-steps.
__global__ void __launch_bounds__(32) k_gemm1(const float* __restrict__ x,
                                              const __bf16* __restrict__ Bp,
                                              const float* __restrict__ bl,
                                              const float* __restrict__ br,
                                              float* __restrict__ xl, float* __restrict__ xr,
                                              int N) {
    int lane = threadIdx.x;
    int kg = lane >> 4, n = lane & 15;
    int row0 = blockIdx.x * 16;
    int rowA = row0 + n;
    if (rowA > N - 1) rowA = N - 1;

    v8f acc8[8];
#pragma unroll
    for (int t = 0; t < 8; t++)
#pragma unroll
        for (int r = 0; r < 8; r++) acc8[t][r] = 0.f;

    const float* arow = x + (size_t)rowA * 256;
#pragma unroll 1
    for (int kk = 0; kk < 8; kk++) {
        float tmp[16];
        const float4* p0 = (const float4*)(arow + kk * 32 + kg * 8);
        const float4* p1 = (const float4*)(arow + kk * 32 + 16 + kg * 8);
        ((float4*)tmp)[0] = p0[0];
        ((float4*)tmp)[1] = p0[1];
        ((float4*)tmp)[2] = p1[0];
        ((float4*)tmp)[3] = p1[1];
        v16bf a;
#pragma unroll
        for (int j = 0; j < 16; j++) a[j] = (__bf16)tmp[j];
        const __bf16* bbase = Bp + (size_t)kk * 8 * 512;
#pragma unroll
        for (int ct = 0; ct < 8; ct++) {
            v16bf b = *(const v16bf*)(bbase + (ct * 32 + lane) * 16);
            acc8[ct] = __builtin_amdgcn_wmma_f32_16x16x32_bf16(
                false, a, false, b, (short)0, acc8[ct], false, false);
        }
    }
#pragma unroll
    for (int ct = 0; ct < 8; ct++) {
        int col = (ct & 3) * 16 + n;
        float bias = (ct < 4) ? bl[col] : br[col];
        float* dstp = (ct < 4) ? xl : xr;
#pragma unroll
        for (int r = 0; r < 8; r++) {
            int row = row0 + r + kg * 8;
            if (row < N) dstp[(size_t)row * 64 + col] = acc8[ct][r] + bias;
        }
    }
}

// GEMM2: h1(N x 64) @ packed(64 x 96) -> xl2(N x 40), xr2(N x 40) (+bias)
__global__ void __launch_bounds__(32) k_gemm2(const float* __restrict__ h1,
                                              const __bf16* __restrict__ Bp,
                                              const float* __restrict__ bl,
                                              const float* __restrict__ br,
                                              float* __restrict__ xl2, float* __restrict__ xr2,
                                              int N) {
    int lane = threadIdx.x;
    int kg = lane >> 4, n = lane & 15;
    int row0 = blockIdx.x * 16;
    int rowA = row0 + n;
    if (rowA > N - 1) rowA = N - 1;

    v8f acc8[6];
#pragma unroll
    for (int t = 0; t < 6; t++)
#pragma unroll
        for (int r = 0; r < 8; r++) acc8[t][r] = 0.f;

    const float* arow = h1 + (size_t)rowA * 64;
#pragma unroll 1
    for (int kk = 0; kk < 2; kk++) {
        float tmp[16];
        const float4* p0 = (const float4*)(arow + kk * 32 + kg * 8);
        const float4* p1 = (const float4*)(arow + kk * 32 + 16 + kg * 8);
        ((float4*)tmp)[0] = p0[0];
        ((float4*)tmp)[1] = p0[1];
        ((float4*)tmp)[2] = p1[0];
        ((float4*)tmp)[3] = p1[1];
        v16bf a;
#pragma unroll
        for (int j = 0; j < 16; j++) a[j] = (__bf16)tmp[j];
        const __bf16* bbase = Bp + (size_t)kk * 6 * 512;
#pragma unroll
        for (int ct = 0; ct < 6; ct++) {
            v16bf b = *(const v16bf*)(bbase + (ct * 32 + lane) * 16);
            acc8[ct] = __builtin_amdgcn_wmma_f32_16x16x32_bf16(
                false, a, false, b, (short)0, acc8[ct], false, false);
        }
    }
#pragma unroll
    for (int ct = 0; ct < 6; ct++) {
        int col = (ct % 3) * 16 + n;
        if (col < 40) {
            float bias = (ct < 3) ? bl[col] : br[col];
            float* dstp = (ct < 3) ? xl2 : xr2;
#pragma unroll
            for (int r = 0; r < 8; r++) {
                int row = row0 + r + kg * 8;
                if (row < N) dstp[(size_t)row * 40 + col] = acc8[ct][r] + bias;
            }
        }
    }
}

// ---------------------------------------------------------------------------
// Layer-1 edge pass A: per (edge, head) logit + atomic max per (dst, head)
__global__ void k_e1a(const float* __restrict__ xl, const float* __restrict__ xr,
                      const int* __restrict__ ei, const float* __restrict__ att,
                      unsigned* __restrict__ mx, int E, int N) {
    int t = blockIdx.x * blockDim.x + threadIdx.x;
    int tot = (E + N) * 8;
    if (t >= tot) return;
    int e = t >> 3, h = t & 7;
    int src = (e < E) ? ei[e] : (e - E);
    int dst = (e < E) ? ei[E + e] : (e - E);
    const float* pl = xl + (size_t)src * 64 + h * 8;
    const float* pr = xr + (size_t)dst * 64 + h * 8;
    const float* pa = att + h * 8;
    float lg = 0.f;
#pragma unroll
    for (int c = 0; c < 8; c++) lg += pa[c] * lrelu(pl[c] + pr[c]);
    atomicMax(mx + (size_t)dst * 8 + h, fenc(lg));
}

// Layer-1 edge pass B: exp weight, accumulate denominator and numerator
__global__ void k_e1b(const float* __restrict__ xl, const float* __restrict__ xr,
                      const int* __restrict__ ei, const float* __restrict__ att,
                      const unsigned* __restrict__ mx, float* __restrict__ den,
                      float* __restrict__ acc, int E, int N) {
    int t = blockIdx.x * blockDim.x + threadIdx.x;
    int tot = (E + N) * 8;
    if (t >= tot) return;
    int e = t >> 3, h = t & 7;
    int src = (e < E) ? ei[e] : (e - E);
    int dst = (e < E) ? ei[E + e] : (e - E);
    const float* pl = xl + (size_t)src * 64 + h * 8;
    const float* pr = xr + (size_t)dst * 64 + h * 8;
    const float* pa = att + h * 8;
    float l[8];
    float lg = 0.f;
#pragma unroll
    for (int c = 0; c < 8; c++) {
        float lv = pl[c];
        l[c] = lv;
        lg += pa[c] * lrelu(lv + pr[c]);
    }
    float w = __expf(lg - fdec(mx[(size_t)dst * 8 + h]));
    atomicAdd(den + (size_t)dst * 8 + h, w);
    float* ap = acc + (size_t)dst * 64 + h * 8;
#pragma unroll
    for (int c = 0; c < 8; c++) atomicAdd(ap + c, w * l[c]);
}

// Layer-1 node finalize: normalize, +bias1, ELU (in place over acc)
__global__ void k_n1(float* __restrict__ acc, const float* __restrict__ den,
                     const float* __restrict__ bias, int N) {
    int t = blockIdx.x * blockDim.x + threadIdx.x;
    if (t >= N * 64) return;
    int nd = t >> 6;
    int hc = t & 63;
    int h = hc >> 3;
    float v = acc[t] / den[(size_t)nd * 8 + h] + bias[hc];
    acc[t] = v > 0.f ? v : (__expf(v) - 1.f);
}

// ---------------------------------------------------------------------------
// Layer-2 edge pass A: per-edge logit (1 head, 40 channels) + atomic max
__global__ void k_e2a(const float* __restrict__ xl2, const float* __restrict__ xr2,
                      const int* __restrict__ ei, const float* __restrict__ att,
                      unsigned* __restrict__ mx, int E, int N) {
    int e = blockIdx.x * blockDim.x + threadIdx.x;
    if (e >= E + N) return;
    int src = (e < E) ? ei[e] : (e - E);
    int dst = (e < E) ? ei[E + e] : (e - E);
    const float* pl = xl2 + (size_t)src * 40;
    const float* pr = xr2 + (size_t)dst * 40;
    float lg = 0.f;
#pragma unroll
    for (int c = 0; c < 40; c++) lg += att[c] * lrelu(pl[c] + pr[c]);
    atomicMax(mx + dst, fenc(lg));
}

// Layer-2 edge pass B
__global__ void k_e2b(const float* __restrict__ xl2, const float* __restrict__ xr2,
                      const int* __restrict__ ei, const float* __restrict__ att,
                      const unsigned* __restrict__ mx, float* __restrict__ den,
                      float* __restrict__ acc, int E, int N) {
    int e = blockIdx.x * blockDim.x + threadIdx.x;
    if (e >= E + N) return;
    int src = (e < E) ? ei[e] : (e - E);
    int dst = (e < E) ? ei[E + e] : (e - E);
    const float* pl = xl2 + (size_t)src * 40;
    const float* pr = xr2 + (size_t)dst * 40;
    float l[40];
    float lg = 0.f;
#pragma unroll
    for (int c = 0; c < 40; c++) {
        float lv = pl[c];
        l[c] = lv;
        lg += att[c] * lrelu(lv + pr[c]);
    }
    float w = __expf(lg - fdec(mx[dst]));
    atomicAdd(den + dst, w);
    float* ap = acc + (size_t)dst * 40;
#pragma unroll
    for (int c = 0; c < 40; c++) atomicAdd(ap + c, w * l[c]);
}

// Layer-2 node finalize: normalize, +bias2, log_softmax -> d_out
__global__ void k_n2(const float* __restrict__ acc, const float* __restrict__ den,
                     const float* __restrict__ bias, float* __restrict__ out, int N) {
    int nd = blockIdx.x * blockDim.x + threadIdx.x;
    if (nd >= N) return;
    float inv = 1.f / den[nd];
    const float* ap = acc + (size_t)nd * 40;
    float v[40];
    float m = -1e30f;
#pragma unroll
    for (int c = 0; c < 40; c++) {
        v[c] = ap[c] * inv + bias[c];
        m = fmaxf(m, v[c]);
    }
    float s = 0.f;
#pragma unroll
    for (int c = 0; c < 40; c++) s += __expf(v[c] - m);
    float lse = m + __logf(s);
    float* op = out + (size_t)nd * 40;
#pragma unroll
    for (int c = 0; c < 40; c++) op[c] = v[c] - lse;
}

// ---------------------------------------------------------------------------
extern "C" void kernel_launch(void* const* d_in, const int* in_sizes, int n_in,
                              void* d_out, int out_size, void* d_ws, size_t ws_size,
                              hipStream_t stream) {
    const float* x    = (const float*)d_in[0];
    const int*   ei   = (const int*)d_in[1];
    const float* W_l1 = (const float*)d_in[2];
    const float* b_l1 = (const float*)d_in[3];
    const float* W_r1 = (const float*)d_in[4];
    const float* b_r1 = (const float*)d_in[5];
    const float* att1 = (const float*)d_in[6];
    const float* bias1 = (const float*)d_in[7];
    const float* W_l2 = (const float*)d_in[8];
    const float* b_l2 = (const float*)d_in[9];
    const float* W_r2 = (const float*)d_in[10];
    const float* b_r2 = (const float*)d_in[11];
    const float* att2 = (const float*)d_in[12];
    const float* bias2 = (const float*)d_in[13];

    int N = in_sizes[0] / 256;
    int E = in_sizes[1] / 2;
    size_t N64 = (size_t)N * 64;

    // workspace layout (f32 units); layer-2 buffers alias layer-1 buffers
    float* ws   = (float*)d_ws;
    float* xl1  = ws;                         // N*64  (reused as xl2: N*40)
    float* xr1  = xl1 + N64;                  // N*64  (reused as xr2: N*40)
    unsigned* mx = (unsigned*)(xr1 + N64);    // N*8   (layer2: first N)
    float* den  = (float*)mx + (size_t)N * 8; // N*8   (layer2: first N)
    float* acc  = den + (size_t)N * 8;        // N*64  (also h1 in-place; layer2: N*40)
    __bf16* Bp1 = (__bf16*)(acc + N64);       // 8*8*32*16
    __bf16* Bp2 = Bp1 + 8 * 8 * 32 * 16;      // 2*6*32*16

    const int B = 256;
    int mb = (N + 15) / 16;
    int zn = N * 80;  // mx + den + acc region
    int e1t = (E + N) * 8;
    int e2t = E + N;

    k_pack1<<<(32768 + B - 1) / B, B, 0, stream>>>(W_l1, W_r1, Bp1);
    k_pack2<<<(6144 + B - 1) / B, B, 0, stream>>>(W_l2, W_r2, Bp2);
    k_zero<<<(zn + B - 1) / B, B, 0, stream>>>(mx, zn);

    k_gemm1<<<mb, 32, 0, stream>>>(x, Bp1, b_l1, b_r1, xl1, xr1, N);
    k_e1a<<<(e1t + B - 1) / B, B, 0, stream>>>(xl1, xr1, ei, att1, mx, E, N);
    k_e1b<<<(e1t + B - 1) / B, B, 0, stream>>>(xl1, xr1, ei, att1, mx, den, acc, E, N);
    k_n1<<<((N * 64) + B - 1) / B, B, 0, stream>>>(acc, den, bias1, N);

    k_gemm2<<<mb, 32, 0, stream>>>(acc, Bp2, b_l2, b_r2, xl1, xr1, N);
    k_zero<<<(zn + B - 1) / B, B, 0, stream>>>(mx, zn);
    k_e2a<<<(e2t + B - 1) / B, B, 0, stream>>>(xl1, xr1, ei, att2, mx, E, N);
    k_e2b<<<(e2t + B - 1) / B, B, 0, stream>>>(xl1, xr1, ei, att2, mx, den, acc, E, N);
    k_n2<<<(N + B - 1) / B, B, 0, stream>>>(acc, den, bias2, (float*)d_out, N);
}
